// MambaAgent_15247133901685
// MI455X (gfx1250) — matrix operand order repaired and checked by reading, not compiled, not run on hardware
//
#include <hip/hip_runtime.h>
#include <hip/hip_bf16.h>

typedef __bf16 bf16_t;
typedef __attribute__((ext_vector_type(16))) __bf16 v16bf;
typedef __attribute__((ext_vector_type(8)))  float  v8f;

#define BATCH   16
#define SEQLEN  1024
#define DMODEL  512
#define DINNER  1024
#define DSTATE  16
#define DTRANK  32
#define DCONV   4
#define ROWS    (BATCH * SEQLEN)   // 16384

// ---------------------------------------------------------------------------
// CDNA5 async LDS-DMA helpers (ASYNCcnt-tracked; see ISA 10.7 / 15.18.3)
__device__ __forceinline__ void async_g2l_b128(unsigned lds_off, const void* g) {
    asm volatile("global_load_async_to_lds_b128 %0, %1, off"
                 :: "v"(lds_off), "v"((unsigned long long)(size_t)g)
                 : "memory");
}
__device__ __forceinline__ void wait_async0() {
    asm volatile("s_wait_asynccnt 0" ::: "memory");
}

// ---------------------------------------------------------------------------
// f32 -> bf16 convert (contiguous)
__global__ void f2bf_kernel(const float* __restrict__ in, bf16_t* __restrict__ out, int n) {
    int i = blockIdx.x * blockDim.x + threadIdx.x;
    if (i < n) out[i] = (bf16_t)in[i];
}

// strided convert: take first `take` cols of a row-major [rows, incols] f32 -> packed bf16
__global__ void f2bf_strided_kernel(const float* __restrict__ in, bf16_t* __restrict__ out,
                                    int rows, int incols, int take) {
    int i = blockIdx.x * blockDim.x + threadIdx.x;
    int n = rows * take;
    if (i >= n) return;
    int r = i / take, c = i % take;
    out[i] = (bf16_t)in[(size_t)r * incols + c];
}

// ---------------------------------------------------------------------------
__device__ __forceinline__ float apply_epi(float x, int EPI) {
    if (EPI == 1) x = x / (1.0f + __expf(-x));               // SiLU
    if (EPI == 2) x = (x > 20.0f) ? x : log1pf(__expf(x));   // softplus
    return x;
}

// Register-blocked WMMA GEMM: out[M,N] = A[M,K](bf16) * W[N,K](bf16)^T (+ epilogue)
// Each wave computes MT*16 rows x NT*16 cols; block = 4 waves stacked in M.
// Requires M % (MT*16*4) == 0, N % (NT*16) == 0, K % 32 == 0.
template <int EPI, typename OT, int MT, int NT>
__global__ void gemm_wmma_tiled(const bf16_t* __restrict__ A, int lda,
                                const bf16_t* __restrict__ W, int ldw,
                                const float* __restrict__ bias,
                                OT* __restrict__ out, int ldo, int K) {
    const int lane = threadIdx.x & 31;
    const int wave = threadIdx.x >> 5;
    const int n0   = blockIdx.x * (NT * 16);
    const int m0   = (blockIdx.y * 4 + wave) * (MT * 16);

    const int h  = lane >> 4;   // half-wave
    const int mr = lane & 15;   // row (A) / col (W) within a tile

    const bf16_t* arow[MT];
    const bf16_t* brow[NT];
#pragma unroll
    for (int mt = 0; mt < MT; ++mt) arow[mt] = A + (size_t)(m0 + mt * 16 + mr) * lda;
#pragma unroll
    for (int nt = 0; nt < NT; ++nt) brow[nt] = W + (size_t)(n0 + nt * 16 + mr) * ldw;

    v8f acc[MT][NT];
#pragma unroll
    for (int mt = 0; mt < MT; ++mt)
#pragma unroll
        for (int nt = 0; nt < NT; ++nt) acc[mt][nt] = (v8f){};

    for (int k0 = 0; k0 < K; k0 += 32) {
        v16bf af[MT], bfr[NT];
#pragma unroll
        for (int mt = 0; mt < MT; ++mt) {
            // A fragment: lanes 0-15: K=k0..k0+15, lanes 16-31: K=k0+16..k0+31 (matches B)
            const bf16_t* ap = arow[mt] + k0 + 16 * h;
#pragma unroll
            for (int j = 0; j < 16; ++j) af[mt][j] = ap[j];
        }
#pragma unroll
        for (int nt = 0; nt < NT; ++nt) {
            // B[k][n] = W[n][k]; lanes 0-15: K=k0..k0+15, lanes 16-31: K=k0+16..k0+31
            const bf16_t* bp = brow[nt] + k0 + 16 * h;
#pragma unroll
            for (int j = 0; j < 16; ++j) bfr[nt][j] = bp[j];
        }
        if (k0 + 32 < K) {
            __builtin_prefetch(arow[0] + k0 + 32, 0, 1);
            __builtin_prefetch(brow[0] + k0 + 32, 0, 1);
        }
#pragma unroll
        for (int mt = 0; mt < MT; ++mt)
#pragma unroll
            for (int nt = 0; nt < NT; ++nt)
                acc[mt][nt] = __builtin_amdgcn_wmma_f32_16x16x32_bf16(
                    false, af[mt], false, bfr[nt], (short)0, acc[mt][nt], false, false);
    }

#pragma unroll
    for (int nt = 0; nt < NT; ++nt) {
        const int col = n0 + nt * 16 + (lane & 15);
        float bv = 0.0f;
        if (EPI != 0) bv = bias[col];
#pragma unroll
        for (int mt = 0; mt < MT; ++mt) {
#pragma unroll
            for (int v = 0; v < 8; ++v) {
                int row = m0 + mt * 16 + 8 * h + v;
                float x = apply_epi(acc[mt][nt][v] + bv, EPI);
                out[(size_t)row * ldo + col] = (OT)x;
            }
        }
    }
}

// Simple one-tile-per-wave WMMA GEMM (for tiny M, e.g. the last-token GEMM)
template <int EPI, typename OT>
__global__ void gemm_wmma_kernel(const bf16_t* __restrict__ A, int lda,
                                 const bf16_t* __restrict__ W, int ldw,
                                 const float* __restrict__ bias,
                                 OT* __restrict__ out, int ldo,
                                 int M, int N, int K) {
    const int lane  = threadIdx.x & 31;
    const int wave  = threadIdx.x >> 5;
    const int tileN = blockIdx.x;
    const int tileM = blockIdx.y * 4 + wave;
    if (tileM * 16 >= M) return;

    const int h  = lane >> 4;
    const int mr = lane & 15;
    const bf16_t* arow = A + (size_t)(tileM * 16 + mr) * lda;
    const bf16_t* brow = W + (size_t)(tileN * 16 + mr) * ldw;

    v8f c = {};
    for (int k0 = 0; k0 < K; k0 += 32) {
        v16bf a, b;
        const bf16_t* ap = arow + k0 + 16 * h;
#pragma unroll
        for (int j = 0; j < 16; ++j) a[j] = ap[j];
        const bf16_t* bp = brow + k0 + 16 * h;
#pragma unroll
        for (int j = 0; j < 16; ++j) b[j] = bp[j];
        c = __builtin_amdgcn_wmma_f32_16x16x32_bf16(false, a, false, b,
                                                    (short)0, c, false, false);
    }
    const int col = tileN * 16 + (lane & 15);
    float bv = 0.0f;
    if (EPI != 0) bv = bias[col];
#pragma unroll
    for (int v = 0; v < 8; ++v) {
        int row = tileM * 16 + 8 * h + v;
        out[(size_t)row * ldo + col] = (OT)apply_epi(c[v] + bv, EPI);
    }
}

// ---------------------------------------------------------------------------
// depthwise causal conv1d over time + SiLU
__global__ void conv_silu_kernel(const bf16_t* __restrict__ xz, const float* __restrict__ cw,
                                 const float* __restrict__ cb, bf16_t* __restrict__ u) {
    int idx = blockIdx.x * blockDim.x + threadIdx.x;
    if (idx >= ROWS * DINNER) return;
    int d = idx % DINNER;
    int l = (idx / DINNER) % SEQLEN;
    int b = idx / (DINNER * SEQLEN);
    float acc = cb[d];
#pragma unroll
    for (int j = 0; j < DCONV; ++j) {
        int ll = l + j - (DCONV - 1);
        if (ll >= 0)
            acc += cw[d * DCONV + j] *
                   (float)xz[((size_t)(b * SEQLEN + ll) * (2 * DINNER)) + d];
    }
    u[idx] = (bf16_t)(acc / (1.0f + __expf(-acc)));
}

// ---------------------------------------------------------------------------
// selective scan: one thread per (b, d); B/C rows staged into LDS via async DMA
#define SCAN_TILE 64
__global__ void scan_kernel(const float* __restrict__ xdbl, const float* __restrict__ delta,
                            const bf16_t* __restrict__ u, const bf16_t* __restrict__ xz,
                            const float* __restrict__ A_log, const float* __restrict__ Dskip,
                            bf16_t* __restrict__ ylast) {
    const int tid = threadIdx.x;
    const int b   = blockIdx.x >> 2;
    const int d   = ((blockIdx.x & 3) << 8) + tid;

    float Ar[DSTATE];
#pragma unroll
    for (int n = 0; n < DSTATE; ++n) Ar[n] = -__expf(A_log[d * DSTATE + n]);
    float h[DSTATE] = {};

    // sBC[lo][0..15] = B row, sBC[lo][16..31] = C row (xdbl cols 32..63, contiguous 128B)
    __shared__ float sBC[SCAN_TILE][2 * DSTATE];

    float yv = 0.0f, ulast = 0.0f;
    for (int l0 = 0; l0 < SEQLEN; l0 += SCAN_TILE) {
        // async-stage SCAN_TILE rows x 128B from global into LDS (16B chunks)
        for (int i = tid; i < SCAN_TILE * 8; i += 256) {
            int lo = i >> 3, c = i & 7;
            const float* g = xdbl + ((size_t)(b * SEQLEN + l0 + lo)) * 64 + 32 + c * 4;
            async_g2l_b128((unsigned)(size_t)&sBC[lo][c * 4], g);
        }
        wait_async0();
        __syncthreads();
        for (int lo = 0; lo < SCAN_TILE; ++lo) {
            int l = l0 + lo;
            size_t r = (size_t)(b * SEQLEN + l);
            float dv = delta[r * DINNER + d];
            float uv = (float)u[r * DINNER + d];
            float du = dv * uv;
#pragma unroll
            for (int n = 0; n < DSTATE; ++n)
                h[n] = __expf(dv * Ar[n]) * h[n] + du * sBC[lo][n];
            if (l == SEQLEN - 1) {
                float y = 0.0f;
#pragma unroll
                for (int n = 0; n < DSTATE; ++n) y += h[n] * sBC[lo][DSTATE + n];
                yv = y; ulast = uv;
            }
        }
        __syncthreads();
    }
    float y = yv + ulast * Dskip[d];
    float z = (float)xz[((size_t)(b * SEQLEN + SEQLEN - 1) * (2 * DINNER)) + DINNER + d];
    y *= z / (1.0f + __expf(-z));
    ylast[b * DINNER + d] = (bf16_t)y;
}

// ---------------------------------------------------------------------------
// layernorm (no affine) + SiLU over one row of 512 (one block per batch)
__global__ void ln_silu_kernel(const float* __restrict__ m, float* __restrict__ nout) {
    const int b = blockIdx.x, tid = threadIdx.x;  // 256 threads
    __shared__ float red[256];
    const float* row = m + (size_t)b * DMODEL;
    float a0 = row[tid], a1 = row[tid + 256];
    red[tid] = a0 + a1;
    __syncthreads();
    for (int o = 128; o > 0; o >>= 1) { if (tid < o) red[tid] += red[tid + o]; __syncthreads(); }
    float mu = red[0] / (float)DMODEL;
    __syncthreads();
    float d0 = a0 - mu, d1 = a1 - mu;
    red[tid] = d0 * d0 + d1 * d1;
    __syncthreads();
    for (int o = 128; o > 0; o >>= 1) { if (tid < o) red[tid] += red[tid + o]; __syncthreads(); }
    float rs = rsqrtf(red[0] / (float)DMODEL + 1e-5f);
    float x0 = d0 * rs, x1 = d1 * rs;
    nout[(size_t)b * DMODEL + tid]       = x0 / (1.0f + __expf(-x0));
    nout[(size_t)b * DMODEL + tid + 256] = x1 / (1.0f + __expf(-x1));
}

// ---------------------------------------------------------------------------
// output heads: action_mean (16x8), action_sd (16x8), value (16)
__global__ void heads_kernel(const float* __restrict__ n,
                             const float* __restrict__ Wc, const float* __restrict__ bc,
                             const float* __restrict__ Wam, const float* __restrict__ bam,
                             const float* __restrict__ Was, const float* __restrict__ bas,
                             float* __restrict__ out) {
    int tid = threadIdx.x;
    if (tid >= BATCH * 17) return;
    int b = tid / 17, o = tid % 17;
    const float* nb = n + (size_t)b * DMODEL;
    if (o < 8) {
        const float* w = Wam + (size_t)o * DMODEL;
        float acc = bam[o];
        for (int k = 0; k < DMODEL; ++k) acc += nb[k] * w[k];
        out[b * 8 + o] = acc;
    } else if (o < 16) {
        int j = o - 8;
        const float* w = Was + (size_t)j * DMODEL;
        float acc = bas[j];
        for (int k = 0; k < DMODEL; ++k) acc += nb[k] * w[k];
        acc = fminf(1.0f, fmaxf(-1.0f, acc));
        out[128 + b * 8 + j] = __expf(acc);
    } else {
        float acc = bc[0];
        for (int k = 0; k < DMODEL; ++k) acc += nb[k] * Wc[k];
        out[256 + b] = acc;
    }
}

// ---------------------------------------------------------------------------
extern "C" void kernel_launch(void* const* d_in, const int* in_sizes, int n_in,
                              void* d_out, int out_size, void* d_ws, size_t ws_size,
                              hipStream_t stream) {
    const float* x       = (const float*)d_in[0];
    const float* W_emb   = (const float*)d_in[1];
    const float* b_emb   = (const float*)d_in[2];
    const float* W_in    = (const float*)d_in[3];
    const float* conv_w  = (const float*)d_in[4];
    const float* conv_b  = (const float*)d_in[5];
    const float* W_xproj = (const float*)d_in[6];
    const float* W_dt    = (const float*)d_in[7];
    const float* b_dt    = (const float*)d_in[8];
    const float* A_log   = (const float*)d_in[9];
    const float* Dskip   = (const float*)d_in[10];
    const float* W_out   = (const float*)d_in[11];
    const float* W_crit  = (const float*)d_in[12];
    const float* b_crit  = (const float*)d_in[13];
    const float* W_amean = (const float*)d_in[14];
    const float* b_amean = (const float*)d_in[15];
    const float* W_astd  = (const float*)d_in[16];
    const float* b_astd  = (const float*)d_in[17];
    float* out = (float*)d_out;

    // workspace carve-up (256B aligned)
    char* base = (char*)d_ws;
    size_t off = 0;
    auto alloc = [&](size_t bytes) -> char* {
        char* p = base + off;
        off = (off + bytes + 255) & ~(size_t)255;
        return p;
    };
    bf16_t* x_bf    = (bf16_t*)alloc((size_t)ROWS * DTRANK * 2);
    bf16_t* wemb_bf = (bf16_t*)alloc((size_t)DMODEL * 32 * 2);
    bf16_t* win_bf  = (bf16_t*)alloc((size_t)2 * DINNER * DMODEL * 2);
    bf16_t* wxp_bf  = (bf16_t*)alloc((size_t)64 * DINNER * 2);
    bf16_t* wdt_bf  = (bf16_t*)alloc((size_t)DINNER * DTRANK * 2);
    bf16_t* wout_bf = (bf16_t*)alloc((size_t)DMODEL * DINNER * 2);
    bf16_t* embed   = (bf16_t*)alloc((size_t)ROWS * DMODEL * 2);
    bf16_t* xz      = (bf16_t*)alloc((size_t)ROWS * 2 * DINNER * 2);
    bf16_t* u       = (bf16_t*)alloc((size_t)ROWS * DINNER * 2);
    float*  xdbl    = (float*) alloc((size_t)ROWS * 64 * 4);
    bf16_t* dt_bf   = (bf16_t*)alloc((size_t)ROWS * DTRANK * 2);
    float*  delta   = (float*) alloc((size_t)ROWS * DINNER * 4);
    bf16_t* ylast   = (bf16_t*)alloc((size_t)BATCH * DINNER * 2);
    float*  mlast   = (float*) alloc((size_t)BATCH * DMODEL * 4);
    float*  nlast   = (float*) alloc((size_t)BATCH * DMODEL * 4);
    (void)ws_size; (void)n_in; (void)in_sizes; (void)out_size;

    auto cdiv = [](int a, int b) { return (a + b - 1) / b; };

    // weight / input conversions to bf16
    f2bf_kernel<<<cdiv(ROWS * 32, 256), 256, 0, stream>>>(x, x_bf, ROWS * 32);
    f2bf_kernel<<<cdiv(DMODEL * 32, 256), 256, 0, stream>>>(W_emb, wemb_bf, DMODEL * 32);
    f2bf_kernel<<<cdiv(2 * DINNER * DMODEL, 256), 256, 0, stream>>>(W_in, win_bf, 2 * DINNER * DMODEL);
    f2bf_kernel<<<cdiv(64 * DINNER, 256), 256, 0, stream>>>(W_xproj, wxp_bf, 64 * DINNER);
    f2bf_kernel<<<cdiv(DINNER * DTRANK, 256), 256, 0, stream>>>(W_dt, wdt_bf, DINNER * DTRANK);
    f2bf_kernel<<<cdiv(DMODEL * DINNER, 256), 256, 0, stream>>>(W_out, wout_bf, DMODEL * DINNER);

    // Register-blocked GEMMs: wave = 32 rows x 64 cols, block = 128 rows x 64 cols
    // GEMM1: embed = silu(x @ W_emb^T + b_emb)   [16384,512] K=32
    gemm_wmma_tiled<1, bf16_t, 2, 4><<<dim3(DMODEL / 64, ROWS / 128), 128, 0, stream>>>(
        x_bf, 32, wemb_bf, 32, b_emb, embed, DMODEL, 32);

    // GEMM2: xz = embed @ W_in^T   [16384,2048] K=512
    gemm_wmma_tiled<0, bf16_t, 2, 4><<<dim3(2 * DINNER / 64, ROWS / 128), 128, 0, stream>>>(
        embed, DMODEL, win_bf, DMODEL, nullptr, xz, 2 * DINNER, DMODEL);

    // depthwise causal conv + SiLU -> u
    conv_silu_kernel<<<cdiv(ROWS * DINNER, 256), 256, 0, stream>>>(xz, conv_w, conv_b, u);

    // GEMM3: x_dbl = u @ W_xproj^T   [16384,64] K=1024  (f32 out for scan precision)
    gemm_wmma_tiled<0, float, 2, 4><<<dim3(64 / 64, ROWS / 128), 128, 0, stream>>>(
        u, DINNER, wxp_bf, DINNER, nullptr, xdbl, 64, DINNER);

    // pack dt (first 32 cols) to bf16 for the delta GEMM
    f2bf_strided_kernel<<<cdiv(ROWS * DTRANK, 256), 256, 0, stream>>>(xdbl, dt_bf, ROWS, 64, DTRANK);

    // GEMM4: delta = softplus(dt @ W_dt^T + b_dt)   [16384,1024] K=32
    gemm_wmma_tiled<2, float, 2, 4><<<dim3(DINNER / 64, ROWS / 128), 128, 0, stream>>>(
        dt_bf, DTRANK, wdt_bf, DTRANK, b_dt, delta, DINNER, DTRANK);

    // selective scan (sequential in L), async LDS staging, fused skip + gate
    scan_kernel<<<BATCH * 4, 256, 0, stream>>>(xdbl, delta, u, xz, A_log, Dskip, ylast);

    // GEMM5: m_last = ylast @ W_out^T   [16,512] K=1024 (tiny M -> simple kernel)
    gemm_wmma_kernel<0, float><<<dim3(DMODEL / 16, 1), 128, 0, stream>>>(
        ylast, DINNER, wout_bf, DINNER, nullptr, mlast, DMODEL, BATCH, DMODEL, DINNER);

    // layernorm + SiLU
    ln_silu_kernel<<<BATCH, 256, 0, stream>>>(mlast, nlast);

    // heads -> d_out = [action_mean(128) | action_sd(128) | value(16)]
    heads_kernel<<<1, 288, 0, stream>>>(nlast, W_crit, b_crit, W_amean, b_amean,
                                        W_astd, b_astd, out);
}
